// WaveletAtt_67843303407763
// MI455X (gfx1250) — matrix operand order repaired and checked by reading, not compiled, not run on hardware
//
#include <hip/hip_runtime.h>
#include <math.h>

typedef __attribute__((ext_vector_type(16))) _Float16 v16h;
typedef __attribute__((ext_vector_type(8)))  _Float16 v8h;
typedef __attribute__((ext_vector_type(8)))  float    v8f;

#define NB      2
#define CD      256
#define NHEADS  8
#define CHD     32
#define H1      64
#define W1      64
#define NP1     (H1*W1)          // 4096
#define H2C     128
#define W2C     128
#define NP2     (H2C*W2C)        // 16384
#define SPLITK  32               // split-K chunks for attn q.k^T
#define CHUNK   (NP2/SPLITK)     // 512 spatial elems per chunk

__device__ __forceinline__ v16h mk16(v8h lo, v8h hi) {
  v16h r;
#pragma unroll
  for (int i = 0; i < 8; ++i) { r[i] = lo[i]; r[i + 8] = hi[i]; }
  return r;
}

// ---------------- f32 -> f16 convert ----------------
__global__ void k_f32_to_f16(const float* __restrict__ a, _Float16* __restrict__ o, size_t n) {
  size_t i = (size_t)blockIdx.x * blockDim.x + threadIdx.x;
  if (i < n) o[i] = (_Float16)a[i];
}

// -------- tiled transpose [B][C][N] f32 -> [B][N][C] f16 (LDS, both sides coalesced) ----
__global__ void k_transpose_f16(const float* __restrict__ X, _Float16* __restrict__ Xt, int Npix) {
  __shared__ float tile[32][33];
  const int b  = blockIdx.z;
  const int n0 = blockIdx.x * 32;
  const int c0 = blockIdx.y * 32;
  const float* Xb = X + (size_t)b * CD * Npix;
  for (int i = threadIdx.y; i < 32; i += 8)
    tile[i][threadIdx.x] = Xb[(size_t)(c0 + i) * Npix + n0 + threadIdx.x];
  __syncthreads();
  _Float16* Ob = Xt + (size_t)b * Npix * CD;
  for (int i = threadIdx.y; i < 32; i += 8)
    Ob[(size_t)(n0 + i) * CD + c0 + threadIdx.x] = (_Float16)tile[threadIdx.x][i];
}

// -------- 1x1-conv GEMM via WMMA: O[b][m][n] = sum_k Wh[m][k] * Xt[b][n][k] --------
__global__ void k_gemm_qkv(const _Float16* __restrict__ Wh,
                           const _Float16* __restrict__ Xt,
                           float* __restrict__ O, int Npix) {
  const int lane  = threadIdx.x & 31;
  const int wave  = threadIdx.x >> 5;
  const int mBase = (blockIdx.y * 8 + wave) * 16;
  const int nBase = blockIdx.x * 16;
  const int b     = blockIdx.z;
  const _Float16* XtB = Xt + (size_t)b * Npix * CD;
  const int row  = mBase + (lane & 15);
  const int col  = nBase + (lane & 15);
  const int klo  = (lane >= 16) ? 16 : 0;  // A: K base per lane half
  const int tklo = (lane < 16) ? 0 : 16;   // B: K base per lane half
  v8f acc = {};
#pragma unroll 4
  for (int k = 0; k < CD; k += 32) {
    const _Float16* ap = Wh + (size_t)row * CD + k + klo;
    v16h a  = *(const v16h*)ap;
    v16h bf = *(const v16h*)(XtB + (size_t)col * CD + k + tklo);
    acc = __builtin_amdgcn_wmma_f32_16x16x32_f16(false, a, false, bf, (short)0, acc, false, false);
  }
  float* Ob = O + (size_t)b * CD * Npix;
  const int mOff = mBase + ((lane >= 16) ? 8 : 0);
#pragma unroll
  for (int r = 0; r < 8; ++r)
    Ob[(size_t)(mOff + r) * Npix + col] = acc[r];
}

// -------- Haar analysis: x[b][c][H][W] -> s[b][c*4+f][H/2][W/2] --------
__global__ void k_haar_dec(const float* __restrict__ x, float* __restrict__ s, int H, int W) {
  int h2 = H >> 1, w2 = W >> 1;
  size_t total = (size_t)NB * CD * h2 * w2;
  size_t idx = (size_t)blockIdx.x * blockDim.x + threadIdx.x;
  if (idx >= total) return;
  int ww = idx % w2; size_t t = idx / w2;
  int hh = t % h2;   t /= h2;
  int c = t % CD;    int b = t / CD;
  const float* xp = x + (((size_t)(b * CD + c)) * H + 2 * hh) * W + 2 * ww;
  float a = xp[0], bb = xp[1], cc = xp[W], dd = xp[W + 1];
  size_t plane = (size_t)h2 * w2;
  float* sp = s + ((size_t)(b * CD + c) * 4) * plane + (size_t)hh * w2 + ww;
  sp[0]         = 0.5f * (a + bb + cc + dd);   // LL
  sp[plane]     = 0.5f * (a + bb - cc - dd);   // hi rows x lo cols
  sp[2 * plane] = 0.5f * (a - bb + cc - dd);   // lo rows x hi cols
  sp[3 * plane] = 0.5f * (a - bb - cc + dd);   // HH
}

// -------- depthwise conv (SAME) + per-channel scale --------
__global__ void k_dwconv(const float* __restrict__ in, const float* __restrict__ w,
                         const float* __restrict__ sc, float* __restrict__ out,
                         int Cn, int H, int W, int ksz) {
  size_t total = (size_t)NB * Cn * H * W;
  size_t idx = (size_t)blockIdx.x * blockDim.x + threadIdx.x;
  if (idx >= total) return;
  int xw = idx % W; size_t t = idx / W;
  int yh = t % H;   t /= H;
  int c = t % Cn;   int b = t / Cn;
  const float* ip = in + ((size_t)(b * Cn + c)) * H * W;
  const float* wp = w + (size_t)c * ksz * ksz;
  int pad = ksz >> 1;
  float acc = 0.f;
  for (int kh = 0; kh < ksz; ++kh) {
    int ih = yh + kh - pad;
    if (ih < 0 || ih >= H) continue;
    for (int kw = 0; kw < ksz; ++kw) {
      int iw = xw + kw - pad;
      if (iw < 0 || iw >= W) continue;
      acc += ip[(size_t)ih * W + iw] * wp[kh * ksz + kw];
    }
  }
  out[idx] = acc * sc[c];
}

// -------- Haar synthesis + fused base depthwise path --------
__global__ void k_wt_synth_base(const float* __restrict__ x, const float* __restrict__ s2,
                                const float* __restrict__ bw, const float* __restrict__ bs,
                                float* __restrict__ y, int H, int W, int ksz) {
  int h2 = H >> 1, w2 = W >> 1;
  size_t total = (size_t)NB * CD * h2 * w2;
  size_t idx = (size_t)blockIdx.x * blockDim.x + threadIdx.x;
  if (idx >= total) return;
  int ww = idx % w2; size_t t = idx / w2;
  int hh = t % h2;   t /= h2;
  int c = t % CD;    int b = t / CD;
  size_t plane = (size_t)h2 * w2;
  const float* sp = s2 + ((size_t)(b * CD + c) * 4) * plane + (size_t)hh * w2 + ww;
  float s0 = sp[0], s1 = sp[plane], s2v = sp[2 * plane], s3 = sp[3 * plane];
  float syn[2][2];
  syn[0][0] = 0.5f * (s0 + s1 + s2v + s3);
  syn[0][1] = 0.5f * (s0 + s1 - s2v - s3);
  syn[1][0] = 0.5f * (s0 - s1 + s2v - s3);
  syn[1][1] = 0.5f * (s0 - s1 - s2v + s3);
  const float* xc = x + ((size_t)(b * CD + c)) * H * W;
  const float* wc = bw + (size_t)c * ksz * ksz;
  float scv = bs[c];
  int pad = ksz >> 1;
  float* yc = y + ((size_t)(b * CD + c)) * H * W;
#pragma unroll
  for (int p = 0; p < 2; ++p)
#pragma unroll
    for (int q = 0; q < 2; ++q) {
      int oh = 2 * hh + p, ow = 2 * ww + q;
      float acc = 0.f;
      for (int kh = 0; kh < ksz; ++kh) {
        int ih = oh + kh - pad;
        if (ih < 0 || ih >= H) continue;
        for (int kw = 0; kw < ksz; ++kw) {
          int iw = ow + kw - pad;
          if (iw < 0 || iw >= W) continue;
          acc += xc[(size_t)ih * W + iw] * wc[kh * ksz + kw];
        }
      }
      yc[(size_t)oh * W + ow] = acc * scv + syn[p][q];
    }
}

// -------- bilinear 64x64 -> 128x128, align_corners=False --------
__global__ void k_resize_q(const float* __restrict__ in, float* __restrict__ out) {
  size_t idx = (size_t)blockIdx.x * blockDim.x + threadIdx.x;
  const size_t total = (size_t)NB * CD * NP2;
  if (idx >= total) return;
  int ow = idx % W2C; size_t t = idx / W2C;
  int oh = t % H2C;   t /= H2C;
  int c = t % CD;     int b = t / CD;
  float sh = (oh + 0.5f) * 0.5f - 0.5f;
  float sw = (ow + 0.5f) * 0.5f - 0.5f;
  int h0 = (int)floorf(sh); float fh = sh - (float)h0;
  int w0 = (int)floorf(sw); float fw = sw - (float)w0;
  int h1c = min(h0 + 1, H1 - 1), w1c = min(w0 + 1, W1 - 1);
  h0 = max(0, min(h0, H1 - 1));  w0 = max(0, min(w0, W1 - 1));
  const float* ip = in + ((size_t)(b * CD + c)) * NP1;
  float v00 = ip[h0 * W1 + w0],  v01 = ip[h0 * W1 + w1c];
  float v10 = ip[h1c * W1 + w0], v11 = ip[h1c * W1 + w1c];
  float top = v00 + (v01 - v00) * fw;
  float bot = v10 + (v11 - v10) * fw;
  out[idx] = top + (bot - top) * fh;
}

// -------- per-row (b, chan) L2 norm over N2 --------
__global__ void k_rownorm(const float* __restrict__ x, float* __restrict__ norms) {
  __shared__ float red[256];
  const float* p = x + (size_t)blockIdx.x * NP2;
  float s = 0.f;
  for (int i = threadIdx.x; i < NP2; i += 256) { float v = p[i]; s += v * v; }
  red[threadIdx.x] = s;
  __syncthreads();
  for (int off = 128; off > 0; off >>= 1) {
    if (threadIdx.x < off) red[threadIdx.x] += red[threadIdx.x + off];
    __syncthreads();
  }
  if (threadIdx.x == 0) norms[blockIdx.x] = fmaxf(sqrtf(red[0]), 1e-12f);
}

__global__ void k_scale_f16(const float* __restrict__ x, const float* __restrict__ norms,
                            _Float16* __restrict__ o) {
  size_t idx = (size_t)blockIdx.x * blockDim.x + threadIdx.x;
  const size_t total = (size_t)NB * CD * NP2;
  if (idx >= total) return;
  size_t row = idx / NP2;
  o[idx] = (_Float16)(x[idx] / norms[row]);
}

// -------- tiled per-head transpose: v[b][h*32+cv][t] -> vT[bh][t][cv] f16 --------
__global__ void k_pack_vT(const float* __restrict__ v, _Float16* __restrict__ vT) {
  __shared__ float tile[32][33];
  const int bh = blockIdx.y;
  const int b = bh >> 3, h = bh & 7;
  const int t0 = blockIdx.x * 32;
  const float* Vb = v + ((size_t)b * CD + h * CHD) * NP2;
  for (int i = threadIdx.y; i < 32; i += 8)
    tile[i][threadIdx.x] = Vb[(size_t)i * NP2 + t0 + threadIdx.x];
  __syncthreads();
  _Float16* Ob = vT + (size_t)bh * NP2 * CHD;
  for (int i = threadIdx.y; i < 32; i += 8)
    Ob[(size_t)(t0 + i) * CHD + threadIdx.x] = (_Float16)tile[threadIdx.x][i];
}

// -------- split-K attn partials: part[bh][s][c][d] = sum_{t in chunk s} q[c][t]*k[d][t] ----
__global__ void k_attn_qk(const _Float16* __restrict__ qn, const _Float16* __restrict__ kn,
                          float* __restrict__ part) {
  const int bh = blockIdx.x;
  const int s  = blockIdx.y;
  const int b = bh / NHEADS, h = bh % NHEADS;
  const int lane = threadIdx.x & 31, wave = threadIdx.x >> 5;  // 4 waves: 2x2 tiles
  const int mBase = (wave >> 1) * 16, nBase = (wave & 1) * 16;
  const _Float16* Q  = qn + ((size_t)b * CD + h * CHD) * NP2;
  const _Float16* Kp = kn + ((size_t)b * CD + h * CHD) * NP2;
  const int row = mBase + (lane & 15);
  const int col = nBase + (lane & 15);
  const int klo = (lane >= 16) ? 16 : 0;
  const int tklo = (lane < 16) ? 0 : 16;
  const int t0 = s * CHUNK;
  v8f acc = {};
#pragma unroll 4
  for (int t = t0; t < t0 + CHUNK; t += 32) {
    const _Float16* ap = Q + (size_t)row * NP2 + t + klo;
    v16h a  = *(const v16h*)ap;
    v16h bf = *(const v16h*)(Kp + (size_t)col * NP2 + t + tklo);
    acc = __builtin_amdgcn_wmma_f32_16x16x32_f16(false, a, false, bf, (short)0, acc, false, false);
  }
  float* P = part + ((size_t)bh * SPLITK + s) * CHD * CHD;
  const int mOff = mBase + ((lane >= 16) ? 8 : 0);
#pragma unroll
  for (int r = 0; r < 8; ++r)
    P[(size_t)(mOff + r) * CHD + col] = acc[r];
}

// -------- reduce split-K partials + temperature + softmax -> f16 --------
__global__ void k_softmax_f16(const float* __restrict__ part, const float* __restrict__ temp,
                              _Float16* __restrict__ out) {
  int row = blockIdx.x * blockDim.x + threadIdx.x;   // row = bh*32 + c
  if (row >= NB * NHEADS * CHD) return;
  const int bh = row >> 5, c = row & 31;
  const int h = bh & 7;
  float a[CHD];
#pragma unroll
  for (int d = 0; d < CHD; ++d) a[d] = 0.f;
  for (int s = 0; s < SPLITK; ++s) {
    const float* pp = part + (((size_t)bh * SPLITK + s) * CHD + c) * CHD;
#pragma unroll
    for (int d = 0; d < CHD; ++d) a[d] += pp[d];
  }
  const float tv = temp[h];
  float m = -1e30f;
#pragma unroll
  for (int d = 0; d < CHD; ++d) { a[d] *= tv; m = fmaxf(m, a[d]); }
  float ssum = 0.f;
#pragma unroll
  for (int d = 0; d < CHD; ++d) { a[d] = __expf(a[d] - m); ssum += a[d]; }
  float inv = 1.f / ssum;
  _Float16* o = out + (size_t)row * CHD;
#pragma unroll
  for (int d = 0; d < CHD; ++d) o[d] = (_Float16)(a[d] * inv);
}

// -------- out0[b][h*32+c][t] = sum_d attn[c][d] * v[d][t] (WMMA, K=32) --------
__global__ void k_attn_v(const _Float16* __restrict__ attn_h, const _Float16* __restrict__ vT,
                         float* __restrict__ out0) {
  const int bh = blockIdx.y;
  const int b = bh >> 3, h = bh & 7;
  const int lane = threadIdx.x & 31, wave = threadIdx.x >> 5;  // 8 waves: 2 m-tiles x 4 n-tiles
  const int mBase = (wave & 1) * 16;
  const int nTile = blockIdx.x * 4 + (wave >> 1);
  const _Float16* A  = attn_h + (size_t)bh * CHD * CHD;
  const _Float16* Vt = vT + (size_t)bh * NP2 * CHD;
  const int row = mBase + (lane & 15);
  const int col = nTile * 16 + (lane & 15);
  const int klo = (lane >= 16) ? 16 : 0;
  const int tklo = (lane < 16) ? 0 : 16;
  const _Float16* ap = A + (size_t)row * CHD + klo;
  v16h a  = *(const v16h*)ap;
  v16h bf = *(const v16h*)(Vt + (size_t)col * CHD + tklo);
  v8f acc = {};
  acc = __builtin_amdgcn_wmma_f32_16x16x32_f16(false, a, false, bf, (short)0, acc, false, false);
  float* Ob = out0 + ((size_t)b * CD + h * CHD) * NP2;
  const int mOff = mBase + ((lane >= 16) ? 8 : 0);
#pragma unroll
  for (int r = 0; r < 8; ++r)
    Ob[(size_t)(mOff + r) * NP2 + col] = acc[r];
}

// =============================== host ===============================
static inline unsigned cdiv(size_t a, unsigned b) { return (unsigned)((a + b - 1) / b); }

static void wtconv_launch(const float* x, float* y,
                          const float* bw, const float* bs,
                          const float* ww, const float* wsc,
                          float* subA, float* subB,
                          int H, int W, int ksz, hipStream_t st) {
  int h2 = H >> 1, w2 = W >> 1;
  size_t nDec = (size_t)NB * CD * h2 * w2;
  k_haar_dec<<<cdiv(nDec, 256), 256, 0, st>>>(x, subA, H, W);
  size_t nConv = (size_t)NB * 4 * CD * h2 * w2;
  k_dwconv<<<cdiv(nConv, 256), 256, 0, st>>>(subA, ww, wsc, subB, 4 * CD, h2, w2, ksz);
  k_wt_synth_base<<<cdiv(nDec, 256), 256, 0, st>>>(x, subB, bw, bs, y, H, W, ksz);
}

extern "C" void kernel_launch(void* const* d_in, const int* in_sizes, int n_in,
                              void* d_out, int out_size, void* d_ws, size_t ws_size,
                              hipStream_t stream) {
  const float* x   = (const float*)d_in[0];
  const float* cf  = (const float*)d_in[1];
  const float* Wq  = (const float*)d_in[2];
  const float* Wk  = (const float*)d_in[3];
  const float* Wv  = (const float*)d_in[4];
  const float* temp = (const float*)d_in[5];
  const float* qdw_bw = (const float*)d_in[6],  *qdw_bs = (const float*)d_in[7];
  const float* qdw_ww = (const float*)d_in[8],  *qdw_ws = (const float*)d_in[9];
  const float* kdw_bw = (const float*)d_in[10], *kdw_bs = (const float*)d_in[11];
  const float* kdw_ww = (const float*)d_in[12], *kdw_ws = (const float*)d_in[13];
  const float* vdw_bw = (const float*)d_in[14], *vdw_bs = (const float*)d_in[15];
  const float* vdw_ww = (const float*)d_in[16], *vdw_ws = (const float*)d_in[17];
  const float* pj_bw  = (const float*)d_in[18], *pj_bs  = (const float*)d_in[19];
  const float* pj_ww  = (const float*)d_in[20], *pj_ws  = (const float*)d_in[21];

  char* p = (char*)d_ws;
  auto carve = [&](size_t bytes) -> void* {
    void* r = (void*)p;
    p += (bytes + 255) & ~(size_t)255;
    return r;
  };
  _Float16* WhQ = (_Float16*)carve((size_t)CD * CD * 2);
  _Float16* WhK = (_Float16*)carve((size_t)CD * CD * 2);
  _Float16* WhV = (_Float16*)carve((size_t)CD * CD * 2);
  _Float16* xT  = (_Float16*)carve((size_t)NB * NP1 * CD * 2);
  _Float16* cfT = (_Float16*)carve((size_t)NB * NP2 * CD * 2);
  float* q0 = (float*)carve((size_t)NB * CD * NP1 * 4);
  float* k0 = (float*)carve((size_t)NB * CD * NP2 * 4);
  float* v0 = (float*)carve((size_t)NB * CD * NP2 * 4);
  float* q1 = (float*)carve((size_t)NB * CD * NP1 * 4);
  float* k1 = (float*)carve((size_t)NB * CD * NP2 * 4);
  float* v1 = (float*)carve((size_t)NB * CD * NP2 * 4);
  float* subA = (float*)carve((size_t)NB * 4 * CD * (NP2 / 4) * 4);
  float* subB = (float*)carve((size_t)NB * 4 * CD * (NP2 / 4) * 4);
  float* rnq = (float*)carve((size_t)NB * CD * 4);
  float* rnk = (float*)carve((size_t)NB * CD * 4);
  _Float16* qn = (_Float16*)carve((size_t)NB * CD * NP2 * 2);
  _Float16* kn = (_Float16*)carve((size_t)NB * CD * NP2 * 2);
  _Float16* vT = (_Float16*)carve((size_t)NB * CD * NP2 * 2);
  float* part = (float*)carve((size_t)NB * NHEADS * SPLITK * CHD * CHD * 4);
  _Float16* attn_h = (_Float16*)carve((size_t)NB * NHEADS * CHD * CHD * 2);
  float* qr   = v0;   // reuse: v0 dead after v1 is produced
  float* out0 = k0;   // reuse: k0 dead after k1 is produced

  // 1) weights -> f16
  size_t wsz = (size_t)CD * CD;
  k_f32_to_f16<<<cdiv(wsz, 256), 256, 0, stream>>>(Wq, WhQ, wsz);
  k_f32_to_f16<<<cdiv(wsz, 256), 256, 0, stream>>>(Wk, WhK, wsz);
  k_f32_to_f16<<<cdiv(wsz, 256), 256, 0, stream>>>(Wv, WhV, wsz);

  // 2) activations -> f16 transposed [b][n][c] (LDS tiled)
  k_transpose_f16<<<dim3(NP1 / 32, CD / 32, NB), dim3(32, 8), 0, stream>>>(x, xT, NP1);
  k_transpose_f16<<<dim3(NP2 / 32, CD / 32, NB), dim3(32, 8), 0, stream>>>(cf, cfT, NP2);

  // 3) 1x1-conv GEMMs (WMMA)
  k_gemm_qkv<<<dim3(NP1 / 16, 2, NB), 256, 0, stream>>>(WhQ, xT, q0, NP1);
  k_gemm_qkv<<<dim3(NP2 / 16, 2, NB), 256, 0, stream>>>(WhK, cfT, k0, NP2);
  k_gemm_qkv<<<dim3(NP2 / 16, 2, NB), 256, 0, stream>>>(WhV, cfT, v0, NP2);

  // 4) WTConv on q, k, v
  wtconv_launch(q0, q1, qdw_bw, qdw_bs, qdw_ww, qdw_ws, subA, subB, H1, W1, 3, stream);
  wtconv_launch(k0, k1, kdw_bw, kdw_bs, kdw_ww, kdw_ws, subA, subB, H2C, W2C, 3, stream);
  wtconv_launch(v0, v1, vdw_bw, vdw_bs, vdw_ww, vdw_ws, subA, subB, H2C, W2C, 3, stream);

  // 5) bilinear upsample q 64->128
  k_resize_q<<<cdiv((size_t)NB * CD * NP2, 256), 256, 0, stream>>>(q1, qr);

  // 6) L2 norms + scale to f16; pack v transposed per head
  k_rownorm<<<NB * CD, 256, 0, stream>>>(qr, rnq);
  k_rownorm<<<NB * CD, 256, 0, stream>>>(k1, rnk);
  k_scale_f16<<<cdiv((size_t)NB * CD * NP2, 256), 256, 0, stream>>>(qr, rnq, qn);
  k_scale_f16<<<cdiv((size_t)NB * CD * NP2, 256), 256, 0, stream>>>(k1, rnk, kn);
  k_pack_vT<<<dim3(NP2 / 32, NB * NHEADS), dim3(32, 8), 0, stream>>>(v1, vT);

  // 7) attention: split-K WMMA partials -> reduce+softmax -> attn*v WMMA
  k_attn_qk<<<dim3(NB * NHEADS, SPLITK), 128, 0, stream>>>(qn, kn, part);
  k_softmax_f16<<<cdiv((size_t)NB * NHEADS * CHD, 256), 256, 0, stream>>>(part, temp, attn_h);
  k_attn_v<<<dim3(NP2 / 64, NB * NHEADS), 256, 0, stream>>>(attn_h, vT, out0);

  // 8) proj WTConv (1x1) -> d_out
  wtconv_launch(out0, (float*)d_out, pj_bw, pj_bs, pj_ww, pj_ws, subA, subB, H2C, W2C, 1, stream);

  (void)in_sizes; (void)n_in; (void)out_size; (void)ws_size;
}
